// ADIOSLoss_712964571804
// MI455X (gfx1250) — compile-verified, matrix operand order checked by reading, not run on hardware
//
#include <hip/hip_runtime.h>
#include <hip/hip_bf16.h>
#include <math.h>

// Problem constants (match reference)
#define B_SZ   4096
#define M_SZ   4
#define D_SZ   512
#define N_SZ   ((M_SZ + 1) * B_SZ)   // 20480 rows in all_emb
#define LDA    520                    // padded LDS row stride (elems): 1040B/row -> 4-bank shift/row

#define ROW_TILES 4                   // 64 rows of A per block, staged in LDS
#define COL_TILES 2                   // 2 column tiles per wave

typedef __bf16 bf16_t;
typedef __bf16  v16bf __attribute__((ext_vector_type(16)));
typedef float   v8f   __attribute__((ext_vector_type(8)));

union FragU {
    v16bf v;
    uint4 q[2];
};

__device__ __forceinline__ float dev_inv_temp(int it) {
    float temp;
    if (it >= 300000) {
        temp = 0.05f;
    } else {
        float prog = (float)it / 300000.0f;
        temp = 0.05f + 0.5f * (0.2f - 0.05f) * (1.0f + cosf(3.14159265358979323846f * prog));
    }
    return 1.0f / temp;
}

// ---------------------------------------------------------------------------
// Kernel 0: zero the denom/pos accumulators (workspace is poisoned by harness)
// ---------------------------------------------------------------------------
__global__ void adios_zero_kernel(float* p, int n) {
    int i = blockIdx.x * blockDim.x + threadIdx.x;
    if (i < n) p[i] = 0.0f;
}

// ---------------------------------------------------------------------------
// Kernel 1: L2-normalize each row of [orig ; masked] and store bf16 all_emb
// One block (256 threads) per row; D = 512 -> 2 floats per thread.
// ---------------------------------------------------------------------------
__global__ void adios_normalize_kernel(const float* __restrict__ orig,
                                       const float* __restrict__ masked,
                                       bf16_t* __restrict__ all_emb) {
    const int row = blockIdx.x;
    const int tid = threadIdx.x;

    const float* src = (row < B_SZ) ? (orig + (size_t)row * D_SZ)
                                    : (masked + (size_t)(row - B_SZ) * D_SZ);

    const float2 v = reinterpret_cast<const float2*>(src)[tid];

    __shared__ float red[256];
    red[tid] = v.x * v.x + v.y * v.y;
    __syncthreads();
    for (int s = 128; s > 0; s >>= 1) {
        if (tid < s) red[tid] += red[tid + s];
        __syncthreads();
    }
    const float scale = 1.0f / fmaxf(sqrtf(red[0]), 1e-12f);

    bf16_t* dst = all_emb + (size_t)row * D_SZ;
    dst[2 * tid + 0] = (bf16_t)(v.x * scale);
    dst[2 * tid + 1] = (bf16_t)(v.y * scale);
}

// ---------------------------------------------------------------------------
// Kernel 2: WMMA sim + fused exp / mask / positive extraction / row reduction
// grid = (80, 64); block = 256 threads (8 waves of 32).
// Block: 64 rows of A (4 row tiles) staged once in LDS (~66.5 KB).
// Wave: 4 row tiles x 2 col tiles = 8 accumulators; K = 512 -> 128 WMMA/wave.
// Per K-step: 8 ds_load_b128 + 4 global_load_b128 -> 8 wmma (1.5 loads/wmma).
// ---------------------------------------------------------------------------
__global__ void __launch_bounds__(256)
adios_sim_kernel(const bf16_t* __restrict__ all_emb,
                 const int* __restrict__ iter_ptr,
                 float* __restrict__ denom,
                 float* __restrict__ pos) {
    __shared__ bf16_t Ald[ROW_TILES * 16 * LDA];   // 64 x 520 bf16 = 66,560 B

    const int tid     = threadIdx.x;
    const int wave    = tid >> 5;
    const int lane    = tid & 31;
    const int l       = lane & 15;     // column within tile / A-row within group
    const int g       = lane >> 4;     // lane group (selects K-halves / row+8)
    const int rowBase = blockIdx.y * (ROW_TILES * 16);

    const float inv_t = dev_inv_temp(*iter_ptr);

    // --- stage 64x512 bf16 A tile into LDS: 4096 uint4 copies ---------------
    {
        const uint4* src = reinterpret_cast<const uint4*>(all_emb + (size_t)rowBase * D_SZ);
        for (int i = tid; i < ROW_TILES * 16 * 64; i += 256) {
            const int r   = i >> 6;        // row 0..63
            const int seg = i & 63;        // 16-byte segment 0..63
            *reinterpret_cast<uint4*>(&Ald[r * LDA + seg * 8]) = src[i];
        }
    }
    __syncthreads();

    // column tiles for this wave
    int colBase[COL_TILES];
    const bf16_t* brow[COL_TILES];
#pragma unroll
    for (int t = 0; t < COL_TILES; ++t) {
        colBase[t] = ((blockIdx.x * 8 + wave) * COL_TILES + t) * 16;
        brow[t]    = all_emb + (size_t)(colBase[t] + l) * D_SZ;
    }

    v8f c[ROW_TILES][COL_TILES];
#pragma unroll
    for (int rt = 0; rt < ROW_TILES; ++rt)
#pragma unroll
        for (int t = 0; t < COL_TILES; ++t)
            c[rt][t] = (v8f){0.f, 0.f, 0.f, 0.f, 0.f, 0.f, 0.f, 0.f};

    // --- K loop: D = 512 in 16 steps of 32 ----------------------------------
#pragma unroll
    for (int kt = 0; kt < 16; ++kt) {
        const int k0 = kt * 32 + g * 8;

        FragU a[ROW_TILES];
#pragma unroll
        for (int rt = 0; rt < ROW_TILES; ++rt) {
            a[rt].q[0] = *reinterpret_cast<const uint4*>(&Ald[(rt * 16 + l) * LDA + k0]);
            a[rt].q[1] = *reinterpret_cast<const uint4*>(&Ald[(rt * 16 + l) * LDA + k0 + 16]);
        }
        FragU b[COL_TILES];
#pragma unroll
        for (int t = 0; t < COL_TILES; ++t) {
            b[t].q[0] = *reinterpret_cast<const uint4*>(brow[t] + k0);
            b[t].q[1] = *reinterpret_cast<const uint4*>(brow[t] + k0 + 16);
        }

#pragma unroll
        for (int rt = 0; rt < ROW_TILES; ++rt)
#pragma unroll
            for (int t = 0; t < COL_TILES; ++t)
                c[rt][t] = __builtin_amdgcn_wmma_f32_16x16x32_bf16(
                               false, a[rt].v, false, b[t].v, (short)0, c[rt][t],
                               false, false);
    }

    // --- fused epilogue (all WMMA done -> divergence is safe now) -----------
    // VGPR r of tile (rt,t) holds (row = rowBase + rt*16 + r + 8*g, col = colBase[t] + l)
#pragma unroll
    for (int rt = 0; rt < ROW_TILES; ++rt) {
#pragma unroll
        for (int r = 0; r < 8; ++r) {
            const int row = rowBase + rt * 16 + r + 8 * g;
            float e = 0.0f;
#pragma unroll
            for (int t = 0; t < COL_TILES; ++t) {
                const int delta = (colBase[t] + l) - row;
                float v = __expf(c[rt][t][r] * inv_t);
                if (delta == 0) {
                    v = 0.0f;                               // self-similarity mask
                } else if ((delta & (B_SZ - 1)) == 0) {
                    atomicAdd(&pos[row], v);                // delta in {B,2B,3B,4B}: positive
                }
                e += v;
            }
            // butterfly sum across the 16 lanes of each half-wave
#pragma unroll
            for (int off = 1; off < 16; off <<= 1)
                e += __shfl_xor(e, off, 32);
            if (l == r)
                atomicAdd(&denom[row], e);
        }
    }
}

// ---------------------------------------------------------------------------
// Kernel 3: loss = mean(-log(pos / (denom + 1e-8)))
// ---------------------------------------------------------------------------
__global__ void adios_finalize_kernel(const float* __restrict__ denom,
                                      const float* __restrict__ pos,
                                      float* __restrict__ out) {
    const int tid = threadIdx.x;
    float s = 0.0f;
    for (int i = tid; i < B_SZ; i += 256)
        s += -logf(pos[i] / (denom[i] + 1e-8f));

    __shared__ float red[256];
    red[tid] = s;
    __syncthreads();
    for (int st = 128; st > 0; st >>= 1) {
        if (tid < st) red[tid] += red[tid + st];
        __syncthreads();
    }
    if (tid == 0) out[0] = red[0] / (float)B_SZ;
}

// ---------------------------------------------------------------------------
extern "C" void kernel_launch(void* const* d_in, const int* in_sizes, int n_in,
                              void* d_out, int out_size, void* d_ws, size_t ws_size,
                              hipStream_t stream) {
    const float* orig   = (const float*)d_in[0];   // [B, D] f32
    const float* masked = (const float*)d_in[1];   // [M, B, D] f32
    const int*   iter   = (const int*)d_in[2];     // scalar

    // workspace layout: bf16 all_emb [N*D] | denom [B] f32 | pos [B] f32
    bf16_t* all_emb = (bf16_t*)d_ws;
    float*  denom   = (float*)((char*)d_ws + (size_t)N_SZ * D_SZ * sizeof(bf16_t));
    float*  pos     = denom + B_SZ;

    // 0) zero accumulators (2*B floats)
    adios_zero_kernel<<<(2 * B_SZ + 255) / 256, 256, 0, stream>>>(denom, 2 * B_SZ);

    // 1) normalize -> bf16 all_emb (20480 rows)
    adios_normalize_kernel<<<N_SZ, 256, 0, stream>>>(orig, masked, all_emb);

    // 2) WMMA sim + fused softmax-denominator/positives
    //    grid.x: 1280 col tiles / (8 waves * 2 tiles) = 80; grid.y: 4096/64 = 64
    dim3 grid(80, B_SZ / (ROW_TILES * 16));
    adios_sim_kernel<<<grid, 256, 0, stream>>>(all_emb, iter, denom, pos);

    // 3) final scalar loss
    adios_finalize_kernel<<<1, 256, 0, stream>>>(denom, pos, (float*)d_out);
}
